// SparseMLPWithLayerSkip_6880537608278
// MI455X (gfx1250) — matrix operand order repaired and checked by reading, not compiled, not run on hardware
//
#include <hip/hip_runtime.h>
#include <cstdint>
#include <cstddef>

// ---- problem dims ----
#define SDIM 8192
#define HDIM 1024
#define EDIM 8
#define IDIM 4096
#define CAP  2048   // capacity = floor(2.0*S/E), rounded even

typedef __attribute__((ext_vector_type(16))) __bf16 v16bf;
typedef __attribute__((ext_vector_type(8)))  float  v8f;

union FragA { unsigned u[8]; v16bf v; };
union FragB { uint4 q[2];    v16bf v; };

static __device__ __forceinline__ unsigned short f2bf(float x) {
  unsigned u = __float_as_uint(x);
  u += 0x7FFFu + ((u >> 16) & 1u);      // round-to-nearest-even
  return (unsigned short)(u >> 16);
}

#define LDS_STRIDE 40   // 128x32 bf16 tile rows padded to 80B (keeps b128 alignment)

// ---- gfx1250 async global->LDS (ASYNCcnt) ----------------------------------
static __device__ __forceinline__ void async_b128(unsigned lds_addr, const void* gaddr) {
  asm volatile("global_load_async_to_lds_b128 %0, %1, off"
               :: "v"(lds_addr), "v"((unsigned long long)(uintptr_t)gaddr)
               : "memory");
}
static __device__ __forceinline__ void wait_async0() {
  asm volatile("s_wait_asynccnt 0x0" ::: "memory");
}

// stage a 128x32 bf16 tile from a row-major [rows][K] bf16 matrix into LDS.
// 256 threads: 2 threads/row, 32B (2x b128) each, via async-to-LDS.
static __device__ __forceinline__ void stage_tile_async(unsigned short* lds,
                                                        const unsigned short* g,
                                                        int K, int k0) {
  const int tid = threadIdx.x;
  const int row = tid >> 1, ch = tid & 1;
  unsigned laddr = (unsigned)(uintptr_t)(const void*)(lds + row * LDS_STRIDE + ch * 16);
  const unsigned short* ga = g + (size_t)row * K + k0 + ch * 16;
  async_b128(laddr, ga);
  async_b128(laddr + 16, ga + 8);
}

// A fragment: 16x32 bf16, lane = M row, interleaved K pairs per ISA 7.12.2.
static __device__ __forceinline__ v16bf load_frag_a(const unsigned short* tile, int rowbase) {
  const int lane = threadIdx.x & 31;
  const int hi = lane >> 4, r = lane & 15;
  const unsigned short* row = tile + (rowbase + r) * LDS_STRIDE;
  FragA f;
#pragma unroll
  for (int i = 0; i < 8; ++i) {
    int k = (i < 4) ? (hi * 8 + 2 * i) : (16 + hi * 8 + 2 * (i - 4));
    f.u[i] = *(const unsigned*)(row + k);
  }
  return f.v;
}

// B fragment: 32x16 bf16, lane = N col; lanes 0-15 K=0..7,16..23, lanes 16-31 K=8..15,24..31.
static __device__ __forceinline__ v16bf load_frag_b(const unsigned short* tile, int rowbase) {
  const int lane = threadIdx.x & 31;
  const int hi = lane >> 4, n = lane & 15;
  const uint4* row = (const uint4*)(tile + (rowbase + n) * LDS_STRIDE + hi * 8);
  FragB f;
  f.q[0] = row[0];
  f.q[1] = row[2];
  return f.v;
}

// ------------------------------------------------------------------
// Kernel 1: per-token LayerNorm + router logits/softmax/top1 + residual init
// ------------------------------------------------------------------
__global__ __launch_bounds__(256) void prep_kernel(
    const float* __restrict__ inp, const float* __restrict__ gate,
    const float* __restrict__ gamma, const float* __restrict__ beta,
    unsigned short* __restrict__ lnbf, int* __restrict__ top1,
    float* __restrict__ prob, float* __restrict__ out) {
  const int s = blockIdx.x;
  const int tid = threadIdx.x;
  const float* x = inp + (size_t)s * HDIM;

  float v[4];
#pragma unroll
  for (int j = 0; j < 4; ++j) v[j] = x[tid * 4 + j];

  __shared__ float red[256];
  __shared__ float s_mu, s_rstd;

  float lsum = v[0] + v[1] + v[2] + v[3];
  red[tid] = lsum;
  __syncthreads();
  for (int off = 128; off > 0; off >>= 1) {
    if (tid < off) red[tid] += red[tid + off];
    __syncthreads();
  }
  if (tid == 0) s_mu = red[0] * (1.0f / HDIM);
  __syncthreads();
  const float mu = s_mu;

  float lsq = 0.0f;
#pragma unroll
  for (int j = 0; j < 4; ++j) { float d = v[j] - mu; lsq += d * d; }
  red[tid] = lsq;
  __syncthreads();
  for (int off = 128; off > 0; off >>= 1) {
    if (tid < off) red[tid] += red[tid + off];
    __syncthreads();
  }
  if (tid == 0) s_rstd = rsqrtf(red[0] * (1.0f / HDIM) + 1e-5f);
  __syncthreads();
  const float rstd = s_rstd;

  float y[4];
#pragma unroll
  for (int j = 0; j < 4; ++j) {
    int h = tid * 4 + j;
    y[j] = (v[j] - mu) * rstd * gamma[h] + beta[h];
  }

  unsigned short* dst = lnbf + (size_t)s * HDIM + tid * 4;
  float* o = out + (size_t)s * HDIM + tid * 4;
#pragma unroll
  for (int j = 0; j < 4; ++j) { dst[j] = f2bf(y[j]); o[j] = v[j]; }

  __shared__ float pr[8 * 256];
#pragma unroll
  for (int e = 0; e < 8; ++e) {
    float p = 0.0f;
#pragma unroll
    for (int j = 0; j < 4; ++j) p += y[j] * gate[e * HDIM + tid * 4 + j];
    pr[e * 256 + tid] = p;
  }
  __syncthreads();
  for (int off = 128; off > 0; off >>= 1) {
    if (tid < off) {
#pragma unroll
      for (int e = 0; e < 8; ++e) pr[e * 256 + tid] += pr[e * 256 + tid + off];
    }
    __syncthreads();
  }
  if (tid == 0) {
    float lmax = -3.4e38f; int am = 0;
#pragma unroll
    for (int e = 0; e < 8; ++e) {
      float l = pr[e * 256];
      if (l > lmax) { lmax = l; am = e; }
    }
    float sum = 0.0f;
#pragma unroll
    for (int e = 0; e < 8; ++e) sum += expf(pr[e * 256] - lmax);
    top1[s] = am;
    prob[s] = 1.0f / sum;
  }
}

// ------------------------------------------------------------------
// Kernel 2: grid-stride fills
// ------------------------------------------------------------------
__global__ void fill_u32_kernel(unsigned* __restrict__ p, int n, unsigned val) {
  int i = blockIdx.x * blockDim.x + threadIdx.x;
  int stride = gridDim.x * blockDim.x;
  for (; i < n; i += stride) p[i] = val;
}

// ------------------------------------------------------------------
// Kernel 3: weight transpose+convert  [K][N] f32 -> [N][K] bf16 (per expert)
// ------------------------------------------------------------------
__global__ __launch_bounds__(256) void transpose_cvt_kernel(
    const float* __restrict__ src, unsigned short* __restrict__ dst, int K, int N) {
  __shared__ unsigned short t[32][33];
  const int e = blockIdx.z;
  src += (size_t)e * K * N;
  dst += (size_t)e * K * N;
  const int nb = blockIdx.x * 32, kb = blockIdx.y * 32;
  const int tx = threadIdx.x & 31, tg = threadIdx.x >> 5;
#pragma unroll
  for (int i = 0; i < 4; ++i) {
    int k = tg * 4 + i;
    t[tx][k] = f2bf(src[(size_t)(kb + k) * N + nb + tx]);  // coalesced read along N
  }
  __syncthreads();
#pragma unroll
  for (int i = 0; i < 4; ++i) {
    int n = tg * 4 + i;
    dst[(size_t)(nb + n) * K + kb + tx] = t[n][tx];        // coalesced write along K
  }
}

// ------------------------------------------------------------------
// Kernel 4: ordered exclusive-cumsum ranking with capacity (single block)
// ------------------------------------------------------------------
__global__ __launch_bounds__(256) void scan_kernel(
    const int* __restrict__ top1, const float* __restrict__ prob,
    float* __restrict__ weight, int* __restrict__ dest, int* __restrict__ inv) {
  __shared__ int eArr[256];
  __shared__ int base[8];
  __shared__ int cnt[8];
  const int tid = threadIdx.x;
  if (tid < 8) base[tid] = 0;
  __syncthreads();
  for (int chunk = 0; chunk < SDIM / 256; ++chunk) {
    int s = chunk * 256 + tid;
    int e = top1[s];
    eArr[tid] = e;
    __syncthreads();
    int r = 0;
    for (int j = 0; j < tid; ++j) r += (eArr[j] == e) ? 1 : 0;
    if (tid < 8) {
      int c = 0;
      for (int j = 0; j < 256; ++j) c += (eArr[j] == tid) ? 1 : 0;
      cnt[tid] = c;
    }
    int rank = base[e] + r;
    bool keep = rank < CAP;
    weight[s] = keep ? prob[s] : 0.0f;
    int d = keep ? (e * CAP + rank) : -1;
    dest[s] = d;
    if (keep) inv[d] = s;
    __syncthreads();
    if (tid < 8) base[tid] += cnt[tid];
    __syncthreads();
  }
}

// ------------------------------------------------------------------
// Kernel 5: dispatch scatter
// ------------------------------------------------------------------
__global__ __launch_bounds__(256) void dispatch_kernel(
    const unsigned short* __restrict__ lnbf, const int* __restrict__ dest,
    unsigned short* __restrict__ disp) {
  const int s = blockIdx.x;
  const int d = dest[s];
  if (d < 0) return;
  const int tid = threadIdx.x;
  const uint2* src = (const uint2*)(lnbf + (size_t)s * HDIM) + tid;
  uint2* dst = (uint2*)(disp + (size_t)d * HDIM) + tid;
  *dst = *src;
}

// ------------------------------------------------------------------
// Kernel 6: GEMM1  Hmid[e] = gelu(X[e] @ Wi[e])   (bf16 WMMA, async double-buffer)
// ------------------------------------------------------------------
__global__ __launch_bounds__(256) void gemm1_kernel(
    const unsigned short* __restrict__ disp, const unsigned short* __restrict__ wiT,
    unsigned short* __restrict__ hmid) {
  const int e  = blockIdx.z;
  const int m0 = blockIdx.y * 128;   // slot rows
  const int n0 = blockIdx.x * 128;   // I cols
  const unsigned short* A = disp + (size_t)e * CAP * HDIM + (size_t)m0 * HDIM;   // [C][K]
  const unsigned short* B = wiT + (size_t)e * IDIM * HDIM + (size_t)n0 * HDIM;   // [N][K]

  __shared__ unsigned short As[2][128 * LDS_STRIDE];
  __shared__ unsigned short Bs[2][128 * LDS_STRIDE];

  const int tid  = threadIdx.x;
  const int wave = tid >> 5;
  const int lane = tid & 31;
  const int wm = (wave & 3) * 32;
  const int wn = (wave >> 2) * 64;

  v8f acc[2][4] = {};

  stage_tile_async(As[0], A, HDIM, 0);
  stage_tile_async(Bs[0], B, HDIM, 0);
  int buf = 0;

  for (int k0 = 0; k0 < HDIM; k0 += 32) {
    wait_async0();
    __syncthreads();
    if (k0 + 32 < HDIM) {  // prefetch next tiles into the other buffer
      stage_tile_async(As[buf ^ 1], A, HDIM, k0 + 32);
      stage_tile_async(Bs[buf ^ 1], B, HDIM, k0 + 32);
    }
    v16bf af[2];
    af[0] = load_frag_a(As[buf], wm);
    af[1] = load_frag_a(As[buf], wm + 16);
    v16bf bfv[4];
#pragma unroll
    for (int ni = 0; ni < 4; ++ni) bfv[ni] = load_frag_b(Bs[buf], wn + ni * 16);
#pragma unroll
    for (int mi = 0; mi < 2; ++mi)
#pragma unroll
      for (int ni = 0; ni < 4; ++ni)
        acc[mi][ni] = __builtin_amdgcn_wmma_f32_16x16x32_bf16(
            false, af[mi], false, bfv[ni], (short)0, acc[mi][ni], false, false);
    buf ^= 1;
  }

  const int hi = lane >> 4, nl = lane & 15;
  const size_t ebase = (size_t)e * CAP * IDIM;
#pragma unroll
  for (int mi = 0; mi < 2; ++mi) {
    int rowb = m0 + wm + mi * 16 + hi * 8;
#pragma unroll
    for (int ni = 0; ni < 4; ++ni) {
      int col = n0 + wn + ni * 16 + nl;
#pragma unroll
      for (int r = 0; r < 8; ++r) {
        float x = acc[mi][ni][r];
        float g = 0.5f * x * (1.0f + erff(x * 0.70710678118654752f));
        hmid[ebase + (size_t)(rowb + r) * IDIM + col] = f2bf(g);
      }
    }
  }
}

// ------------------------------------------------------------------
// Kernel 7: GEMM2  out[tok] = inp[tok] + w[tok] * (Hmid[e] @ Wo[e])[slot]
// ------------------------------------------------------------------
__global__ __launch_bounds__(256) void gemm2_kernel(
    const unsigned short* __restrict__ hmid, const unsigned short* __restrict__ woT,
    const int* __restrict__ inv, const float* __restrict__ weight,
    const float* __restrict__ inp, float* __restrict__ out) {
  const int e  = blockIdx.z;
  const int m0 = blockIdx.y * 128;   // slot rows
  const int n0 = blockIdx.x * 128;   // H cols
  const unsigned short* A = hmid + (size_t)e * CAP * IDIM + (size_t)m0 * IDIM;   // [C][K]
  const unsigned short* B = woT + (size_t)e * HDIM * IDIM + (size_t)n0 * IDIM;   // [N][K]

  __shared__ unsigned short As[2][128 * LDS_STRIDE];
  __shared__ unsigned short Bs[2][128 * LDS_STRIDE];

  const int tid  = threadIdx.x;
  const int wave = tid >> 5;
  const int lane = tid & 31;
  const int wm = (wave & 3) * 32;
  const int wn = (wave >> 2) * 64;

  v8f acc[2][4] = {};

  stage_tile_async(As[0], A, IDIM, 0);
  stage_tile_async(Bs[0], B, IDIM, 0);
  int buf = 0;

  for (int k0 = 0; k0 < IDIM; k0 += 32) {
    wait_async0();
    __syncthreads();
    if (k0 + 32 < IDIM) {
      stage_tile_async(As[buf ^ 1], A, IDIM, k0 + 32);
      stage_tile_async(Bs[buf ^ 1], B, IDIM, k0 + 32);
    }
    v16bf af[2];
    af[0] = load_frag_a(As[buf], wm);
    af[1] = load_frag_a(As[buf], wm + 16);
    v16bf bfv[4];
#pragma unroll
    for (int ni = 0; ni < 4; ++ni) bfv[ni] = load_frag_b(Bs[buf], wn + ni * 16);
#pragma unroll
    for (int mi = 0; mi < 2; ++mi)
#pragma unroll
      for (int ni = 0; ni < 4; ++ni)
        acc[mi][ni] = __builtin_amdgcn_wmma_f32_16x16x32_bf16(
            false, af[mi], false, bfv[ni], (short)0, acc[mi][ni], false, false);
    buf ^= 1;
  }

  const int hi = lane >> 4, nl = lane & 15;
#pragma unroll
  for (int mi = 0; mi < 2; ++mi) {
    int rowb = m0 + wm + mi * 16 + hi * 8;
#pragma unroll
    for (int ni = 0; ni < 4; ++ni) {
      int col = n0 + wn + ni * 16 + nl;
#pragma unroll
      for (int r = 0; r < 8; ++r) {
        int c = rowb + r;
        int s = inv[e * CAP + c];
        if (s >= 0) {
          float w = weight[s];
          size_t idx = (size_t)s * HDIM + col;
          out[idx] = inp[idx] + w * acc[mi][ni][r];
        }
      }
    }
  }
}

// ------------------------------------------------------------------
// host launcher
// ------------------------------------------------------------------
extern "C" void kernel_launch(void* const* d_in, const int* in_sizes, int n_in,
                              void* d_out, int out_size, void* d_ws, size_t ws_size,
                              hipStream_t stream) {
  (void)in_sizes; (void)n_in; (void)out_size; (void)ws_size;
  const float* inp   = (const float*)d_in[0];
  const float* gate  = (const float*)d_in[1];
  const float* gamma = (const float*)d_in[2];
  const float* beta  = (const float*)d_in[3];
  const float* wi    = (const float*)d_in[4];
  const float* wo    = (const float*)d_in[5];
  float* out = (float*)d_out;

  char* ws = (char*)d_ws;
  size_t off = 0;
  unsigned short* lnbf = (unsigned short*)(ws + off); off += (size_t)SDIM * HDIM * 2;        // 16 MB
  unsigned short* disp = (unsigned short*)(ws + off); off += (size_t)EDIM * CAP * HDIM * 2;  // 32 MB
  unsigned short* hmid = (unsigned short*)(ws + off); off += (size_t)EDIM * CAP * IDIM * 2;  // 128 MB
  unsigned short* wiT  = (unsigned short*)(ws + off); off += (size_t)EDIM * HDIM * IDIM * 2; // 64 MB
  unsigned short* woT  = (unsigned short*)(ws + off); off += (size_t)EDIM * IDIM * HDIM * 2; // 64 MB
  int*   top1   = (int*)(ws + off);   off += (size_t)SDIM * 4;
  float* prob   = (float*)(ws + off); off += (size_t)SDIM * 4;
  float* weight = (float*)(ws + off); off += (size_t)SDIM * 4;
  int*   dest   = (int*)(ws + off);   off += (size_t)SDIM * 4;
  int*   inv    = (int*)(ws + off);   off += (size_t)EDIM * CAP * 4;

  // 1) LN + router + residual init
  prep_kernel<<<SDIM, 256, 0, stream>>>(inp, gate, gamma, beta, lnbf, top1, prob, out);
  // 2) weight transpose+convert to bf16 [N][K] (once per launch)
  transpose_cvt_kernel<<<dim3(IDIM / 32, HDIM / 32, EDIM), 256, 0, stream>>>(wi, wiT, HDIM, IDIM);
  transpose_cvt_kernel<<<dim3(HDIM / 32, IDIM / 32, EDIM), 256, 0, stream>>>(wo, woT, IDIM, HDIM);
  // 3) zero dispatch buffer, inv = -1
  fill_u32_kernel<<<4096, 256, 0, stream>>>((unsigned*)disp, EDIM * CAP * HDIM / 2, 0u);
  fill_u32_kernel<<<64, 256, 0, stream>>>((unsigned*)inv, EDIM * CAP, 0xFFFFFFFFu);
  // 4) capacity-constrained ranking
  scan_kernel<<<1, 256, 0, stream>>>(top1, prob, weight, dest, inv);
  // 5) dispatch scatter
  dispatch_kernel<<<SDIM, 256, 0, stream>>>(lnbf, dest, disp);
  // 6) expert GEMM1 + gelu
  gemm1_kernel<<<dim3(IDIM / 128, CAP / 128, EDIM), 256, 0, stream>>>(disp, wiT, hmid);
  // 7) expert GEMM2 + fused combine
  gemm2_kernel<<<dim3(HDIM / 128, CAP / 128, EDIM), 256, 0, stream>>>(hmid, woT, inv, weight, inp, out);
}
